// HistogramLoss_266287972996
// MI455X (gfx1250) — matrix operand order repaired and checked by reading, not compiled, not run on hardware
//
#include <hip/hip_runtime.h>
#include <hip/hip_bf16.h>
#include <math.h>

#define NBINS 201
#define WAVES_PER_BLOCK 8
#define THREADS (WAVES_PER_BLOCK * 32)
#define NSUB (THREADS / 8)      // one histogram copy per 8-lane group
#define MAIN_BLOCKS 2048

typedef _Float16 v16h __attribute__((ext_vector_type(16)));
typedef _Float16 v8h  __attribute__((ext_vector_type(8)));
typedef float    v8f  __attribute__((ext_vector_type(8)));

// ---------------------------------------------------------------------------
// fp32 -> fp16 conversion (inputs are L2-normalized, |val| <= 1, f16 is safe)
// ---------------------------------------------------------------------------
__global__ void cvt_f32_to_f16(const float* __restrict__ in,
                               _Float16* __restrict__ out, int n) {
  int i = blockIdx.x * blockDim.x + threadIdx.x;
  if (i < n) out[i] = (_Float16)in[i];
}

// ---------------------------------------------------------------------------
// zero the global accumulators (hist[2*NBINS] floats + 2 uint counters)
// ---------------------------------------------------------------------------
__global__ void zero_ws(float* __restrict__ hist, unsigned* __restrict__ cnt) {
  for (int b = threadIdx.x; b < 2 * NBINS; b += blockDim.x) hist[b] = 0.0f;
  if (threadIdx.x < 2) cnt[threadIdx.x] = 0u;
}

// ---------------------------------------------------------------------------
// Main: 2x2-tile-blocked WMMA GEMM + in-register soft-binning into
// per-8-lane-group LDS histograms.
//   sims[i][j] = dot(x1[i], x2[j]);  pair valid iff i < j
//   pos iff t2[i] * t1[j] > 0
// ---------------------------------------------------------------------------
__global__ __launch_bounds__(THREADS)
void hist_main(const _Float16* __restrict__ x1h,
               const _Float16* __restrict__ x2h,
               const float* __restrict__ t1,
               const float* __restrict__ t2,
               float* __restrict__ g_hist,      // [2*NBINS] (pos | neg)
               unsigned* __restrict__ g_cnt,    // [2]       (pos, neg)
               int d, int nTilesDim)
{
  __shared__ float lhist[NSUB][2 * NBINS];
  __shared__ unsigned scnt[2];

  const float stepv    = 2.0f / (float)(NBINS - 1);
  const float inv_step = 1.0f / stepv;

  // zero LDS
  for (int b = threadIdx.x; b < NSUB * 2 * NBINS; b += THREADS)
    (&lhist[0][0])[b] = 0.0f;
  if (threadIdx.x < 2) scnt[threadIdx.x] = 0u;
  __syncthreads();

  const int lane = threadIdx.x & 31;
  const int wave = threadIdx.x >> 5;
  const int m    = lane & 15;   // row (A) / col (B,C,D) within 16x16 tile
  const int hi   = lane >> 4;   // lane half-group
  float* const hw = &lhist[threadIdx.x >> 3][0];

  unsigned cp = 0, cn = 0;

  // per-element soft binning straight from an accumulator register block
  auto binTile = [&](const v8f& c, int ibase, int j,
                     const float* t2v, float t1j) {
    #pragma unroll
    for (int v = 0; v < 8; ++v) {
      const int i = ibase + v;
      if (i < j) {
        const float s   = c[v];
        const bool  pos = (t2v[v] * t1j) > 0.0f;
        int idx = (int)floorf((s + 1.0f) * inv_step);
        idx = idx < 0 ? 0 : (idx > NBINS - 1 ? NBINS - 1 : idx);
        const float delta = (float)idx * stepv - 1.0f;
        const float frac  = (s - delta) * inv_step;
        int idx_hi = idx + 1;
        if (idx_hi > NBINS - 1) idx_hi = NBINS - 1;
        const int base = pos ? 0 : NBINS;
        atomicAdd(hw + base + idx,    1.0f - frac);   // ds_add_f32
        atomicAdd(hw + base + idx_hi, frac);
        if (pos) ++cp; else ++cn;
      }
    }
  };

  const int T2       = nTilesDim >> 1;          // 32x32 macro-tile grid
  const int totMacro = T2 * T2;
  const int gwave    = blockIdx.x * WAVES_PER_BLOCK + wave;
  const int nwaves   = gridDim.x * WAVES_PER_BLOCK;
  const int ksteps   = d >> 5;                  // K chunks of 32

  for (int t = gwave; t < totMacro; t += nwaves) {
    const int TI = t / T2;
    const int TJ = t - TI * T2;
    if (TJ < TI) continue;                      // macro-tile below diagonal

    const int ti0 = 2 * TI, ti1 = 2 * TI + 1;
    const int tj0 = 2 * TJ, tj1 = 2 * TJ + 1;

    const _Float16* arow0 = x1h + (size_t)(ti0 * 16 + m) * (size_t)d;
    const _Float16* arow1 = x1h + (size_t)(ti1 * 16 + m) * (size_t)d;
    const _Float16* brow0 = x2h + (size_t)(tj0 * 16 + m) * (size_t)d;
    const _Float16* brow1 = x2h + (size_t)(tj1 * 16 + m) * (size_t)d;

    v8f c00 = {}, c01 = {}, c10 = {}, c11 = {};
    for (int kk = 0; kk < ksteps; ++kk) {
      // A fragments (16x32 f16, M = m): VGPR0-3 <- K = 16*hi+0..7,
      //                                 VGPR4-7 <- K = 16*hi+8..15
      v8h a0lo = *(const v8h*)(arow0 + kk * 32 + 16 * hi);
      v8h a0hi = *(const v8h*)(arow0 + kk * 32 + 16 * hi + 8);
      v8h a1lo = *(const v8h*)(arow1 + kk * 32 + 16 * hi);
      v8h a1hi = *(const v8h*)(arow1 + kk * 32 + 16 * hi + 8);
      v16h a0 = __builtin_shufflevector(a0lo, a0hi,
                  0,1,2,3,4,5,6,7,8,9,10,11,12,13,14,15);
      v16h a1 = __builtin_shufflevector(a1lo, a1hi,
                  0,1,2,3,4,5,6,7,8,9,10,11,12,13,14,15);
      // B fragments (32x16 f16, N = m): lanes 0-15 K=0..15, 16-31 K=16..31
      // B[k][n] = x2[n][k] => 16 contiguous halfs of x2 row n
      v16h b0 = *(const v16h*)(brow0 + kk * 32 + 16 * hi);
      v16h b1 = *(const v16h*)(brow1 + kk * 32 + 16 * hi);
      c00 = __builtin_amdgcn_wmma_f32_16x16x32_f16(
                false, a0, false, b0, (short)0, c00, false, false);
      c01 = __builtin_amdgcn_wmma_f32_16x16x32_f16(
                false, a0, false, b1, (short)0, c01, false, false);
      c10 = __builtin_amdgcn_wmma_f32_16x16x32_f16(
                false, a1, false, b0, (short)0, c10, false, false);
      c11 = __builtin_amdgcn_wmma_f32_16x16x32_f16(
                false, a1, false, b1, (short)0, c11, false, false);
    }

    // targets: two column values, two 8-row blocks
    const float t1j0 = t1[tj0 * 16 + m];
    const float t1j1 = t1[tj1 * 16 + m];
    const float4* tp0 = (const float4*)(t2 + ti0 * 16 + 8 * hi);
    const float4* tp1 = (const float4*)(t2 + ti1 * 16 + 8 * hi);
    const float4 p0a = tp0[0], p0b = tp0[1];
    const float4 p1a = tp1[0], p1b = tp1[1];
    const float t2v0[8] = {p0a.x, p0a.y, p0a.z, p0a.w,
                           p0b.x, p0b.y, p0b.z, p0b.w};
    const float t2v1[8] = {p1a.x, p1a.y, p1a.z, p1a.w,
                           p1b.x, p1b.y, p1b.z, p1b.w};

    const int ib0 = ti0 * 16 + 8 * hi;
    const int ib1 = ti1 * 16 + 8 * hi;
    const int j0  = tj0 * 16 + m;
    const int j1  = tj1 * 16 + m;

    binTile(c00, ib0, j0, t2v0, t1j0);
    binTile(c01, ib0, j1, t2v0, t1j1);
    binTile(c10, ib1, j0, t2v1, t1j0);   // all-dead on diagonal macro-tiles
    binTile(c11, ib1, j1, t2v1, t1j1);
  }

  atomicAdd(&scnt[0], cp);
  atomicAdd(&scnt[1], cn);
  __syncthreads();

  // reduce per-group LDS histograms -> global
  for (int b = threadIdx.x; b < 2 * NBINS; b += THREADS) {
    float s = 0.0f;
    #pragma unroll
    for (int w = 0; w < NSUB; ++w) s += lhist[w][b];
    atomicAdd(&g_hist[b], s);
  }
  if (threadIdx.x == 0) {
    atomicAdd(&g_cnt[0], scnt[0]);
    atomicAdd(&g_cnt[1], scnt[1]);
  }
}

// ---------------------------------------------------------------------------
// finalize: normalize, CDF of pos hist, dot with neg hist -> scalar loss
// ---------------------------------------------------------------------------
__global__ void finalize(const float* __restrict__ g_hist,
                         const unsigned* __restrict__ g_cnt,
                         float* __restrict__ out) {
  if (threadIdx.x == 0 && blockIdx.x == 0) {
    float cp = (float)g_cnt[0]; if (cp < 1.0f) cp = 1.0f;
    float cn = (float)g_cnt[1]; if (cn < 1.0f) cn = 1.0f;
    float cdf = 0.0f, loss = 0.0f;
    for (int k = 0; k < NBINS; ++k) {
      cdf  += g_hist[k] / cp;
      loss += (g_hist[NBINS + k] / cn) * cdf;
    }
    out[0] = loss;
  }
}

// ---------------------------------------------------------------------------
extern "C" void kernel_launch(void* const* d_in, const int* in_sizes, int n_in,
                              void* d_out, int out_size, void* d_ws, size_t ws_size,
                              hipStream_t stream) {
  const float* x1 = (const float*)d_in[0];
  const float* x2 = (const float*)d_in[1];
  const float* t1 = (const float*)d_in[2];
  const float* t2 = (const float*)d_in[3];

  const int n = in_sizes[2];            // N = 8192
  const int d = in_sizes[0] / n;        // D = 128
  const size_t nelem = (size_t)n * (size_t)d;

  // workspace layout: x1h (f16) | x2h (f16) | hist[2*NBINS] f32 | cnt[2] u32
  char* ws = (char*)d_ws;
  _Float16* x1h   = (_Float16*)ws;
  _Float16* x2h   = (_Float16*)(ws + nelem * sizeof(_Float16));
  float*    ghist = (float*)(ws + 2 * nelem * sizeof(_Float16));
  unsigned* gcnt  = (unsigned*)(ghist + 2 * NBINS);

  const int cvtBlocks = (int)((nelem + 255) / 256);
  cvt_f32_to_f16<<<cvtBlocks, 256, 0, stream>>>(x1, x1h, (int)nelem);
  cvt_f32_to_f16<<<cvtBlocks, 256, 0, stream>>>(x2, x2h, (int)nelem);
  zero_ws<<<1, 256, 0, stream>>>(ghist, gcnt);

  const int nTilesDim = n / 16;         // 512 (assumed even; N % 32 == 0)
  hist_main<<<MAIN_BLOCKS, THREADS, 0, stream>>>(x1h, x2h, t1, t2,
                                                 ghist, gcnt, d, nTilesDim);
  finalize<<<1, 32, 0, stream>>>(ghist, gcnt, (float*)d_out);
}
